// SuperResolution_28784870818538
// MI455X (gfx1250) — matrix operand order repaired and checked by reading, not compiled, not run on hardware
//
#include <hip/hip_runtime.h>
#include <cstddef>
#include <cstdint>

typedef __attribute__((ext_vector_type(16))) __bf16   v16bf;
typedef __attribute__((ext_vector_type(8)))  float    v8f;
typedef __attribute__((ext_vector_type(4)))  unsigned int v4u;

__device__ __forceinline__ float bf2f(unsigned short h) {
    unsigned int u = ((unsigned int)h) << 16;
    return __builtin_bit_cast(float, u);
}
__device__ __forceinline__ unsigned short f2bf(float f) {
    unsigned int u = __builtin_bit_cast(unsigned int, f);
    unsigned int r = u + 0x7FFFu + ((u >> 16) & 1u);   // round-to-nearest-even
    return (unsigned short)(r >> 16);
}

// ---------------------------------------------------------------------------
// Kernel 1: w_vec (positional encoding of rays[0,:3,0,0]) + all affine styles.
// Style region (floats): s0@0(32), s1@32(256), st0@288(256), s2@544(256),
//                        s3@800(128), st1@928(128)  -> 1056 floats total.
// ---------------------------------------------------------------------------
__global__ void style_kernel(const float* __restrict__ rays,
                             const float* a0w, const float* a0b,
                             const float* a1w, const float* a1b,
                             const float* t0aw, const float* t0ab,
                             const float* a2w, const float* a2b,
                             const float* a3w, const float* a3b,
                             const float* t1aw, const float* t1ab,
                             float* __restrict__ st) {
    __shared__ float wv[72];
    int t = threadIdx.x;
    if (t < 36) {
        float p  = rays[(t / 12) * 16384];          // rays[0, t/12, 0, 0]
        float pt = p * (float)(1 << (t % 12));
        wv[t]      = sinf(pt);
        wv[36 + t] = cosf(pt);
    }
    __syncthreads();
    const float inv72 = 0.1178511301977579f;        // 1/sqrt(72)
    for (int idx = t; idx < 1056; idx += blockDim.x) {
        const float *aw, *ab; int j, off; float scale = 1.0f;
        if      (idx < 32)  { aw = a0w;  ab = a0b;  j = idx;       off = 0;   }
        else if (idx < 288) { aw = a1w;  ab = a1b;  j = idx - 32;  off = 32;  }
        else if (idx < 544) { aw = t0aw; ab = t0ab; j = idx - 288; off = 288; scale = 0.0625f; }
        else if (idx < 800) { aw = a2w;  ab = a2b;  j = idx - 544; off = 544; }
        else if (idx < 928) { aw = a3w;  ab = a3b;  j = idx - 800; off = 800; }
        else                { aw = t1aw; ab = t1ab; j = idx - 928; off = 928; scale = 0.08838834764831845f; }
        float acc = 0.f;
        #pragma unroll 8
        for (int k = 0; k < 72; ++k) acc += wv[k] * aw[j * 72 + k];
        st[off + j] = (acc * inv72 + ab[j]) * scale;
    }
}

// ---------------------------------------------------------------------------
// Kernel 2: 1x1 conv (3->32) + ReLU + BN. channels-last bf16 feat + fp32 rgb.
// ---------------------------------------------------------------------------
__global__ void feat_kernel(const float* __restrict__ rgbs,
                            const float* __restrict__ cw, const float* __restrict__ cb,
                            const float* __restrict__ g,  const float* __restrict__ be,
                            const float* __restrict__ mu, const float* __restrict__ var,
                            unsigned short* __restrict__ feat, float* __restrict__ rgbout) {
    int pix = blockIdx.x * blockDim.x + threadIdx.x;
    if (pix >= 16384) return;
    float r0 = rgbs[pix], r1 = rgbs[16384 + pix], r2 = rgbs[32768 + pix];
    for (int co = 0; co < 32; ++co) {
        float v = cw[co*3]*r0 + cw[co*3+1]*r1 + cw[co*3+2]*r2 + cb[co];
        v = fmaxf(v, 0.f);
        v = (v - mu[co]) * rsqrtf(var[co] + 1e-5f) * g[co] + be[co];
        feat[(size_t)pix * 32 + co] = f2bf(v);
        if (co < 3) rgbout[pix * 3 + co] = v;
    }
}

// ---------------------------------------------------------------------------
// Kernel 3: antialiased 2x upsample, channels-last bf16 -> bf16.
// out[2i] = 0.25*x[i-1] + 0.75*x[i] ; out[2i+1] = 0.75*x[i] + 0.25*x[i+1]
// ---------------------------------------------------------------------------
__global__ void upsample2x_bf16(const unsigned short* __restrict__ src,
                                unsigned short* __restrict__ dst,
                                int Hs, int Ws, int C) {
    long long idx = (long long)blockIdx.x * blockDim.x + threadIdx.x;
    long long total = 4LL * Hs * Ws * C;
    if (idx >= total) return;
    int c = (int)(idx % C);
    long long t = idx / C;
    int w = (int)(t % (2 * Ws));
    int h = (int)(t / (2 * Ws));
    int hi0, hi1; float hw0, hw1;
    if (h & 1) { hi0 = h >> 1;       hw0 = 0.75f; hi1 = (h >> 1) + 1; hw1 = 0.25f; }
    else       { hi0 = (h >> 1) - 1; hw0 = 0.25f; hi1 = h >> 1;       hw1 = 0.75f; }
    int wi0, wi1; float ww0, ww1;
    if (w & 1) { wi0 = w >> 1;       ww0 = 0.75f; wi1 = (w >> 1) + 1; ww1 = 0.25f; }
    else       { wi0 = (w >> 1) - 1; ww0 = 0.25f; wi1 = w >> 1;       ww1 = 0.75f; }
    float v00 = (hi0 >= 0 && hi0 < Hs && wi0 >= 0 && wi0 < Ws) ? bf2f(src[((size_t)hi0*Ws+wi0)*C + c]) : 0.f;
    float v01 = (hi0 >= 0 && hi0 < Hs && wi1 >= 0 && wi1 < Ws) ? bf2f(src[((size_t)hi0*Ws+wi1)*C + c]) : 0.f;
    float v10 = (hi1 >= 0 && hi1 < Hs && wi0 >= 0 && wi0 < Ws) ? bf2f(src[((size_t)hi1*Ws+wi0)*C + c]) : 0.f;
    float v11 = (hi1 >= 0 && hi1 < Hs && wi1 >= 0 && wi1 < Ws) ? bf2f(src[((size_t)hi1*Ws+wi1)*C + c]) : 0.f;
    float acc = hw0*(ww0*v00 + ww1*v01) + hw1*(ww0*v10 + ww1*v11);
    dst[((size_t)h * 2 * Ws + w) * C + c] = f2bf(acc);
}

// ---------------------------------------------------------------------------
// Kernel 4: modulate + demodulate + repack weights to bf16 [Co][tap(9)][Ci].
// ---------------------------------------------------------------------------
__global__ void prep_w_kernel(const float* __restrict__ wsrc,
                              const float* __restrict__ s,
                              unsigned short* __restrict__ wdst,
                              int Ci, int demod) {
    int co = blockIdx.x;
    int K  = Ci * 9;
    const float* wrow = wsrc + (size_t)co * K;
    __shared__ float red[256];
    __shared__ float dsh;
    float acc = 0.f;
    for (int idx = threadIdx.x; idx < K; idx += 256) {
        int ci = idx / 9, tap = idx % 9;
        float v = wrow[ci * 9 + tap] * s[ci];
        acc += v * v;
    }
    red[threadIdx.x] = acc; __syncthreads();
    for (int o = 128; o > 0; o >>= 1) {
        if ((int)threadIdx.x < o) red[threadIdx.x] += red[threadIdx.x + o];
        __syncthreads();
    }
    if (threadIdx.x == 0) dsh = demod ? rsqrtf(red[0] + 1e-8f) : 1.0f;
    __syncthreads();
    float d = dsh;
    for (int idx = threadIdx.x; idx < K; idx += 256) {
        int ci = idx / 9, tap = idx % 9;
        wdst[(size_t)co * K + tap * Ci + ci] = f2bf(wrow[ci * 9 + tap] * s[ci] * d);
    }
}

// ---------------------------------------------------------------------------
// Kernel 5: 3x3 modconv as implicit GEMM on v_wmma_f32_16x16x32_bf16.
//
// Block = 128 threads (4 waves) -> 64(co) x 32(px) macro-tile, one image row.
// Stage the activation halo patch [3 rows][34 cols][Ci] into LDS with
// global_load_async_to_lds_b128 (ASYNCcnt), zero-fill halo with ds stores.
// Each wave: 16co x 32px = two accumulators sharing one A (weight) fragment.
// B fragments come from ds_load_b128 out of the shared patch.
// Epilogue: +bias, leaky_relu(0.2)*sqrt(2), clip +-256, bf16 ch-last store.
// ---------------------------------------------------------------------------
__global__ __launch_bounds__(128)
void modconv3x3_wmma(const unsigned short* __restrict__ act,
                     const unsigned short* __restrict__ wp,
                     const float* __restrict__ bias,
                     unsigned short* __restrict__ out,
                     int H, int W, int Ci, int Co) {
    extern __shared__ unsigned short sbuf[];   // [3][34][Ci]

    const int tid  = threadIdx.x;
    const int wv   = tid >> 5;          // wave 0..3
    const int l    = tid & 31;
    const int half = l >> 4;
    const int lm   = l & 15;
    const int w0   = blockIdx.x * 32;
    const int h    = blockIdx.y;
    const int co0  = blockIdx.z * 64 + wv * 16;

    // ---- stage activation patch: rows h-1..h+1, cols w0-1..w0+32, all Ci ----
    {
        const int chunks = Ci >> 3;              // 16B (8 bf16) transfers
        const int total  = 3 * 34 * chunks;
        const v4u zero = {0u, 0u, 0u, 0u};
        for (int t = tid; t < total; t += 128) {
            int chunk = t % chunks;
            int col   = (t / chunks) % 34;
            int row   = t / (chunks * 34);
            int hh = h + row - 1;
            int ww = w0 + col - 1;
            unsigned short* lp = sbuf + ((size_t)(row * 34 + col) * Ci + chunk * 8);
            if (hh >= 0 && hh < H && ww >= 0 && ww < W) {
                const unsigned short* gp = act + ((size_t)hh * W + ww) * Ci + chunk * 8;
                unsigned laddr = (unsigned)(uintptr_t)lp;
                unsigned long long gaddr = (unsigned long long)(uintptr_t)gp;
                asm volatile("global_load_async_to_lds_b128 %0, %1, off"
                             :: "v"(laddr), "v"(gaddr) : "memory");
            } else {
                *(v4u*)lp = zero;                // zero halo
            }
        }
        asm volatile("s_wait_asynccnt 0x0" ::: "memory");
        __syncthreads();
    }

    v8f c0 = {0.f,0.f,0.f,0.f,0.f,0.f,0.f,0.f};
    v8f c1 = {0.f,0.f,0.f,0.f,0.f,0.f,0.f,0.f};

    const unsigned short* wrow = wp + (size_t)(co0 + lm) * 9 * Ci;

    for (int tap = 0; tap < 9; ++tap) {
        const int tr = tap / 3, tc = tap % 3;
        const unsigned short* arow = wrow + tap * Ci;
        if (tap < 8) __builtin_prefetch(wrow + (tap + 1) * Ci, 0, 1);
        // LDS row base for this tap; B cols: sub*16 + lm + tc
        const unsigned short* b0row = sbuf + ((size_t)(tr * 34 + lm + tc) * Ci);
        const unsigned short* b1row = b0row + (size_t)16 * Ci;
        for (int cb = 0; cb < Ci; cb += 32) {
            union { v4u u[2]; v16bf v; } A, B0, B1;
            // A lane layout: elems 0..15 -> K = cb + half*16 + e, row = lm
            A.u[0] = *(const v4u*)(arow + cb + half * 16);
            A.u[1] = *(const v4u*)(arow + cb + half * 16 + 8);
            // B lane layout: elems 0..15 -> K = cb + half*16 + e, col = lm
            B0.u[0] = *(const v4u*)(b0row + cb + half * 16);
            B0.u[1] = *(const v4u*)(b0row + cb + half * 16 + 8);
            B1.u[0] = *(const v4u*)(b1row + cb + half * 16);
            B1.u[1] = *(const v4u*)(b1row + cb + half * 16 + 8);
            c0 = __builtin_amdgcn_wmma_f32_16x16x32_bf16(
                     false, A.v, false, B0.v, (short)0, c0, false, false);
            c1 = __builtin_amdgcn_wmma_f32_16x16x32_bf16(
                     false, A.v, false, B1.v, (short)0, c1, false, false);
        }
    }

    const float s2 = 1.4142135623730951f;
    #pragma unroll
    for (int r = 0; r < 8; ++r) {
        int co = co0 + r + half * 8;                 // C/D layout: M = r + 8*(l>>4)
        float b = bias[co];
        float v0 = c0[r] + b;
        v0 = (v0 < 0.f ? 0.2f * v0 : v0) * s2;
        v0 = fminf(fmaxf(v0, -256.f), 256.f);
        out[((size_t)h * W + w0 + lm) * Co + co] = f2bf(v0);
        float v1 = c1[r] + b;
        v1 = (v1 < 0.f ? 0.2f * v1 : v1) * s2;
        v1 = fminf(fmaxf(v1, -256.f), 256.f);
        out[((size_t)h * W + w0 + 16 + lm) * Co + co] = f2bf(v1);
    }
}

// ---------------------------------------------------------------------------
// Kernel 6: to_rgb (1x1 modulated conv, no demod) + fused fp32 2x upsample of
// the skip image + clip. Optional planar (NCHW) output for the final store.
// ---------------------------------------------------------------------------
__global__ void to_rgb_kernel(const unsigned short* __restrict__ act,
                              const float* __restrict__ tw, const float* __restrict__ tb,
                              const float* __restrict__ s,
                              const float* __restrict__ imgsrc,   // [H/2][W/2][3]
                              float* __restrict__ outp,
                              int H, int W, int Ci, int planar) {
    int pix = blockIdx.x * blockDim.x + threadIdx.x;
    if (pix >= H * W) return;
    int h = pix / W, w = pix % W;

    float a0 = 0.f, a1 = 0.f, a2 = 0.f;
    const unsigned short* arow = act + (size_t)pix * Ci;
    for (int ci = 0; ci < Ci; ++ci) {
        float xv = bf2f(arow[ci]) * s[ci];
        a0 += xv * tw[ci];
        a1 += xv * tw[Ci + ci];
        a2 += xv * tw[2 * Ci + ci];
    }

    int Hs = H >> 1, Ws = W >> 1;
    int hi0, hi1; float hw0, hw1;
    if (h & 1) { hi0 = h >> 1;       hw0 = 0.75f; hi1 = (h >> 1) + 1; hw1 = 0.25f; }
    else       { hi0 = (h >> 1) - 1; hw0 = 0.25f; hi1 = h >> 1;       hw1 = 0.75f; }
    int wi0, wi1; float ww0, ww1;
    if (w & 1) { wi0 = w >> 1;       ww0 = 0.75f; wi1 = (w >> 1) + 1; ww1 = 0.25f; }
    else       { wi0 = (w >> 1) - 1; ww0 = 0.25f; wi1 = w >> 1;       ww1 = 0.75f; }

    float up[3];
    #pragma unroll
    for (int ch = 0; ch < 3; ++ch) {
        float v00 = (hi0 >= 0 && hi0 < Hs && wi0 >= 0 && wi0 < Ws) ? imgsrc[((size_t)hi0*Ws+wi0)*3 + ch] : 0.f;
        float v01 = (hi0 >= 0 && hi0 < Hs && wi1 >= 0 && wi1 < Ws) ? imgsrc[((size_t)hi0*Ws+wi1)*3 + ch] : 0.f;
        float v10 = (hi1 >= 0 && hi1 < Hs && wi0 >= 0 && wi0 < Ws) ? imgsrc[((size_t)hi1*Ws+wi0)*3 + ch] : 0.f;
        float v11 = (hi1 >= 0 && hi1 < Hs && wi1 >= 0 && wi1 < Ws) ? imgsrc[((size_t)hi1*Ws+wi1)*3 + ch] : 0.f;
        up[ch] = hw0 * (ww0 * v00 + ww1 * v01) + hw1 * (ww0 * v10 + ww1 * v11);
    }

    float o0 = up[0] + fminf(fmaxf(a0 + tb[0], -256.f), 256.f);
    float o1 = up[1] + fminf(fmaxf(a1 + tb[1], -256.f), 256.f);
    float o2 = up[2] + fminf(fmaxf(a2 + tb[2], -256.f), 256.f);
    if (planar) {
        outp[pix] = o0; outp[(size_t)H*W + pix] = o1; outp[2*(size_t)H*W + pix] = o2;
    } else {
        outp[pix*3] = o0; outp[pix*3+1] = o1; outp[pix*3+2] = o2;
    }
}

// ---------------------------------------------------------------------------
extern "C" void kernel_launch(void* const* d_in, const int* in_sizes, int n_in,
                              void* d_out, int out_size, void* d_ws, size_t ws_size,
                              hipStream_t stream) {
    (void)in_sizes; (void)n_in; (void)out_size; (void)ws_size;
    const float* rays   = (const float*)d_in[0];
    const float* rgbs   = (const float*)d_in[1];
    const float* conv_w = (const float*)d_in[2];
    const float* conv_b = (const float*)d_in[3];
    const float* bn_g   = (const float*)d_in[4];
    const float* bn_b   = (const float*)d_in[5];
    const float* bn_m   = (const float*)d_in[6];
    const float* bn_v   = (const float*)d_in[7];
    const float* a0_w = (const float*)d_in[8];  const float* a0_b = (const float*)d_in[9];
    const float* w0   = (const float*)d_in[10]; const float* b0   = (const float*)d_in[11];
    const float* a1_w = (const float*)d_in[12]; const float* a1_b = (const float*)d_in[13];
    const float* w1   = (const float*)d_in[14]; const float* b1   = (const float*)d_in[15];
    const float* t0aw = (const float*)d_in[16]; const float* t0ab = (const float*)d_in[17];
    const float* t0_w = (const float*)d_in[18]; const float* t0_b = (const float*)d_in[19];
    const float* a2_w = (const float*)d_in[20]; const float* a2_b = (const float*)d_in[21];
    const float* w2   = (const float*)d_in[22]; const float* b2   = (const float*)d_in[23];
    const float* a3_w = (const float*)d_in[24]; const float* a3_b = (const float*)d_in[25];
    const float* w3   = (const float*)d_in[26]; const float* b3   = (const float*)d_in[27];
    const float* t1aw = (const float*)d_in[28]; const float* t1ab = (const float*)d_in[29];
    const float* t1_w = (const float*)d_in[30]; const float* t1_b = (const float*)d_in[31];

    // ---- workspace carving (deterministic) ----
    size_t off = 0;
    auto alloc = [&](size_t bytes) { size_t r = off; off += (bytes + 255) & ~(size_t)255; return r; };
    char* ws = (char*)d_ws;
    size_t oStyles = alloc(1056 * 4);
    size_t oWp0  = alloc((size_t)256 * 288  * 2);
    size_t oWp1  = alloc((size_t)256 * 2304 * 2);
    size_t oWp2  = alloc((size_t)128 * 2304 * 2);
    size_t oWp3  = alloc((size_t)128 * 1152 * 2);
    size_t oFeat = alloc((size_t)16384 * 32 * 2);          // 128x128x32 bf16
    size_t oRgb  = alloc((size_t)16384 * 3 * 4);           // 128x128x3  f32
    size_t oUp0  = alloc((size_t)65536 * 32 * 2);          // 256x256x32 bf16
    size_t oX01  = alloc((size_t)65536 * 256 * 2 * 2);     // x0 + x1 (64 MB), x2 aliases it
    size_t oImg  = alloc((size_t)65536 * 3 * 4);           // 256x256x3  f32
    size_t oUp2  = alloc((size_t)262144 * 256 * 2);        // 512x512x256 bf16 (128MB), x3 aliases
    size_t oX0 = oX01;
    size_t oX1 = oX01 + (size_t)65536 * 256 * 2;
    size_t oX2 = oX01;               // x0/x1 dead by then (64 MB needed)
    size_t oX3 = oUp2;               // up2 dead by then

    float* st = (float*)(ws + oStyles);
    float* s0  = st + 0;   float* s1  = st + 32;  float* st0 = st + 288;
    float* s2v = st + 544; float* s3v = st + 800; float* st1 = st + 928;
    unsigned short* wp0 = (unsigned short*)(ws + oWp0);
    unsigned short* wp1 = (unsigned short*)(ws + oWp1);
    unsigned short* wp2 = (unsigned short*)(ws + oWp2);
    unsigned short* wp3 = (unsigned short*)(ws + oWp3);
    unsigned short* feat = (unsigned short*)(ws + oFeat);
    float* rgb128 = (float*)(ws + oRgb);
    unsigned short* up0 = (unsigned short*)(ws + oUp0);
    unsigned short* x0  = (unsigned short*)(ws + oX0);
    unsigned short* x1  = (unsigned short*)(ws + oX1);
    unsigned short* x2  = (unsigned short*)(ws + oX2);
    unsigned short* x3  = (unsigned short*)(ws + oX3);
    unsigned short* up2 = (unsigned short*)(ws + oUp2);
    float* img256 = (float*)(ws + oImg);
    float* outp   = (float*)d_out;

    auto blocks = [](long long n, int b) { return (unsigned)((n + b - 1) / b); };
    auto ldsBytes = [](int Ci) { return (size_t)3 * 34 * Ci * 2; };

    // 1) styles
    style_kernel<<<1, 128, 0, stream>>>(rays, a0_w, a0_b, a1_w, a1_b, t0aw, t0ab,
                                        a2_w, a2_b, a3_w, a3_b, t1aw, t1ab, st);
    // 2) feat + rgb
    feat_kernel<<<blocks(16384, 256), 256, 0, stream>>>(rgbs, conv_w, conv_b,
                                                        bn_g, bn_b, bn_m, bn_v, feat, rgb128);
    // 3) upsample feat 128->256 (32 ch)
    upsample2x_bf16<<<blocks(4LL*128*128*32, 256), 256, 0, stream>>>(feat, up0, 128, 128, 32);
    // 4) layer0: 32 -> 256 @ 256x256
    prep_w_kernel<<<256, 256, 0, stream>>>(w0, s0, wp0, 32, 1);
    modconv3x3_wmma<<<dim3(256/32, 256, 256/64), 128, ldsBytes(32), stream>>>(
        up0, wp0, b0, x0, 256, 256, 32, 256);
    // 5) layer1: 256 -> 256 @ 256x256
    prep_w_kernel<<<256, 256, 0, stream>>>(w1, s1, wp1, 256, 1);
    modconv3x3_wmma<<<dim3(256/32, 256, 256/64), 128, ldsBytes(256), stream>>>(
        x0, wp1, b1, x1, 256, 256, 256, 256);
    // 6) img256 = up(rgb128) + to_rgb0(x1)
    to_rgb_kernel<<<blocks(256LL*256, 256), 256, 0, stream>>>(x1, t0_w, t0_b, st0,
                                                              rgb128, img256, 256, 256, 256, 0);
    // 7) upsample x1 256->512 (256 ch)
    upsample2x_bf16<<<blocks(4LL*256*256*256, 256), 256, 0, stream>>>(x1, up2, 256, 256, 256);
    // 8) layer2: 256 -> 128 @ 512x512
    prep_w_kernel<<<128, 256, 0, stream>>>(w2, s2v, wp2, 256, 1);
    modconv3x3_wmma<<<dim3(512/32, 512, 128/64), 128, ldsBytes(256), stream>>>(
        up2, wp2, b2, x2, 512, 512, 256, 128);
    // 9) layer3: 128 -> 128 @ 512x512
    prep_w_kernel<<<128, 256, 0, stream>>>(w3, s3v, wp3, 128, 1);
    modconv3x3_wmma<<<dim3(512/32, 512, 128/64), 128, ldsBytes(128), stream>>>(
        x2, wp3, b3, x3, 512, 512, 128, 128);
    // 10) d_out = up(img256) + to_rgb1(x3)   (planar NCHW fp32)
    to_rgb_kernel<<<blocks(512LL*512, 256), 256, 0, stream>>>(x3, t1_w, t1_b, st1,
                                                              img256, outp, 512, 512, 128, 1);
}
